// LogBayes_60335700574661
// MI455X (gfx1250) — compile-verified, matrix operand order checked by reading, not run on hardware
//
#include <hip/hip_runtime.h>
#include <hip/hip_bf16.h>

#define B_SZ 8
#define T_SZ 4096
#define DIN 256
#define H_SZ 1024
#define DOUT 256
#define NLAYER 2

#define SCH 512            // scan chunk length
#define SNC (T_SZ / SCH)   // 8 chunks per (b,h)

typedef __attribute__((ext_vector_type(16))) __bf16 v16bf;
typedef __attribute__((ext_vector_type(8)))  float  v8f;

__device__ __forceinline__ unsigned short f32_to_bf16(float f) {
  unsigned int u = __float_as_uint(f);
  u += 0x7fffu + ((u >> 16) & 1u);   // round-to-nearest-even
  return (unsigned short)(u >> 16);
}

// pack two fp32 -> packed bf16x2 with a single v_perm_b32
__device__ __forceinline__ unsigned pack2_bf16(float x, float y) {
  unsigned ux = __float_as_uint(x);
  ux += 0x7fffu + ((ux >> 16) & 1u);
  unsigned uy = __float_as_uint(y);
  uy += 0x7fffu + ((uy >> 16) & 1u);
  return __builtin_amdgcn_perm(uy, ux, 0x07060302u);
}

union FragU { v16bf v; uint4 q[2]; };

// ---------------------------------------------------------------------------
// Generic bf16 WMMA GEMM:  C[M,N] = act( A[M,K] * W[N,K]^T + bias[N] )
// Block tile 128x128, BK=32, 256 threads = 8 waves (wave32).
// Wave (wm,wn) in 2x4 grid owns a 64x32 region = 4x2 WMMA 16x16 tiles.
// Double-buffered LDS, one barrier per K-step. Register discipline for the
// CU-mode ~112-VGPR cap: W tile is copied global->LDS at the top of the
// iteration (short live range; its wait is partial, leaving A's loads
// outstanding); A tile is register-staged (8 packed VGPRs) and stored after
// the WMMAs; A fragments transient, only B fragments resident.
// ---------------------------------------------------------------------------
template <bool A_F32, bool RELU, bool OUT_F32>
__global__ __launch_bounds__(256) void gemm_wmma_kernel(
    const void* __restrict__ Ap, int lda,
    const unsigned short* __restrict__ W,
    const float* __restrict__ bias,
    void* __restrict__ Cp, int ldc, int K) {
  __shared__ unsigned short As[2][128 * 40];  // +8 halves pad, 16B-aligned rows
  __shared__ unsigned short Bs[2][128 * 40];
  const int BUFE = 128 * 40;  // elements per buffer

  const int tid  = threadIdx.x;
  const int lane = tid & 31;
  const int wid  = tid >> 5;
  const int wm   = wid & 1;        // 0..1 -> 64-row slab
  const int wn   = wid >> 1;       // 0..3 -> 32-col slab
  const int m0   = blockIdx.y * 128;
  const int n0   = blockIdx.x * 128;

  v8f acc[4][2];
#pragma unroll
  for (int i = 0; i < 4; ++i)
#pragma unroll
    for (int j = 0; j < 2; ++j)
#pragma unroll
      for (int e = 0; e < 8; ++e) acc[i][j][e] = 0.0f;

  const int lrow = tid >> 1;         // 0..127 tile row loaded by this thread
  const int lcol = (tid & 1) * 16;   // 0 or 16 (half of the 32-wide K slice)

  // loop-invariant row-base pointers (only +k0 varies per K-step)
  const float*          aF = (const float*)Ap + (size_t)(m0 + lrow) * lda + lcol;
  const unsigned short* aB =
      (const unsigned short*)Ap + (size_t)(m0 + lrow) * lda + lcol;
  const unsigned short* wP = W + (size_t)(n0 + lrow) * K + lcol;
  unsigned short* as_dst = &As[0][lrow * 40 + lcol];
  unsigned short* bs_dst = &Bs[0][lrow * 40 + lcol];

  // staged A tile (8 VGPRs, either path)
  uint2 pa[4];  // fp32 path: packed at load time
  uint4 ra[2];  // bf16 path

  auto load_A = [&](int k0) {
    if (A_F32) {
#pragma unroll
      for (int j = 0; j < 4; ++j) {
        float4 v = *(const float4*)(aF + k0 + 4 * j);
        pa[j].x = pack2_bf16(v.x, v.y);
        pa[j].y = pack2_bf16(v.z, v.w);
      }
    } else {
      ra[0] = *(const uint4*)(aB + k0);
      ra[1] = *(const uint4*)(aB + k0 + 8);
    }
  };
  auto store_A = [&](int buf) {
    if (A_F32) {
#pragma unroll
      for (int j = 0; j < 4; ++j)
        *(uint2*)(as_dst + buf * BUFE + 4 * j) = pa[j];
    } else {
      *(uint4*)(as_dst + buf * BUFE)     = ra[0];
      *(uint4*)(as_dst + buf * BUFE + 8) = ra[1];
    }
  };

  // prologue: stage K-step 0 into buffer 0
  {
    uint4 w0 = *(const uint4*)(wP);
    uint4 w1 = *(const uint4*)(wP + 8);
    load_A(0);
    *(uint4*)(bs_dst)     = w0;
    *(uint4*)(bs_dst + 8) = w1;
    store_A(0);
  }
  __syncthreads();

  const int nk  = K >> 5;
  int       cur = 0;

  // fragment gather bases per ISA 16-bit 16x32 layout:
  // lane<16: K 0..7 in elems 0..7, K 16..23 in elems 8..15 (row/col = lane)
  // lane>=16: K 8..15 / 24..31                      (row/col = lane-16)
  const int kb   = (lane >> 4) << 3;  // 0 or 8
  const int rsel = lane & 15;
  const unsigned short* as_rd = &As[0][(wm * 64 + rsel) * 40 + kb];
  const unsigned short* bs_rd = &Bs[0][(wn * 32 + rsel) * 40 + kb];

#pragma unroll 1
  for (int ki = 0; ki < nk; ++ki) {
    if (ki + 1 < nk) {
      const int kn = (ki + 1) << 5;
      // W loads FIRST, then A loads: the wait before W's ds_store is partial
      // (LOADcnt decrements in order), so A's loads stay outstanding.
      uint4 w0 = *(const uint4*)(wP + kn);
      uint4 w1 = *(const uint4*)(wP + kn + 8);
      load_A(kn);
      *(uint4*)(bs_dst + (cur ^ 1) * BUFE)     = w0;  // safe after prev barrier
      *(uint4*)(bs_dst + (cur ^ 1) * BUFE + 8) = w1;
    }

    // B fragments resident (16 VGPRs); A fragments transient (8 at a time).
    FragU bfr[2];
#pragma unroll
    for (int nt = 0; nt < 2; ++nt) {
      bfr[nt].q[0] = *(const uint4*)(bs_rd + cur * BUFE + nt * 16 * 40);
      bfr[nt].q[1] = *(const uint4*)(bs_rd + cur * BUFE + nt * 16 * 40 + 16);
    }
#pragma unroll
    for (int mt = 0; mt < 4; ++mt) {
      FragU af;
      af.q[0] = *(const uint4*)(as_rd + cur * BUFE + mt * 16 * 40);
      af.q[1] = *(const uint4*)(as_rd + cur * BUFE + mt * 16 * 40 + 16);
      acc[mt][0] = __builtin_amdgcn_wmma_f32_16x16x32_bf16(
          false, af.v, false, bfr[0].v, (short)0, acc[mt][0], false, false);
      acc[mt][1] = __builtin_amdgcn_wmma_f32_16x16x32_bf16(
          false, af.v, false, bfr[1].v, (short)0, acc[mt][1], false, false);
    }

    if (ki + 1 < nk) {
      store_A(cur ^ 1);   // A's global loads have completed during the WMMAs
      __syncthreads();    // publish both tiles for the next iteration
      cur ^= 1;
    }
  }

  // ---- epilogue: C/D layout — VGPR r, lane<16 -> M=r, lane>=16 -> M=r+8 ----
  const int rhi = (lane >> 4) << 3;
#pragma unroll
  for (int mt = 0; mt < 4; ++mt) {
#pragma unroll
    for (int nt = 0; nt < 2; ++nt) {
      int col  = n0 + wn * 32 + nt * 16 + rsel;
      float bv = bias[col];
#pragma unroll
      for (int r = 0; r < 8; ++r) {
        int row = m0 + wm * 64 + mt * 16 + rhi + r;
        float v = acc[mt][nt][r] + bv;
        if (RELU) v = (v > 0.0f) ? v : 0.01f * v;
        if (OUT_F32)
          ((float*)Cp)[(size_t)row * ldc + col] = v;
        else
          ((unsigned short*)Cp)[(size_t)row * ldc + col] = f32_to_bf16(v);
      }
    }
  }
}

// ---------------------------------------------------------------------------
// fp32 -> bf16 elementwise conversion (weights)
// ---------------------------------------------------------------------------
__global__ void cvt_bf16_kernel(const float* __restrict__ in,
                                unsigned int* __restrict__ out, int n2) {
  int i = blockIdx.x * 256 + threadIdx.x;  // one packed pair per thread
  if (i < n2) {
    float2 v = ((const float2*)in)[i];
    out[i] = pack2_bf16(v.x, v.y);
  }
}

// ---------------------------------------------------------------------------
// Segmented reset-scan (log-Bayes accumulation) over T per (b,h), 3 passes.
// Monoid: combine(a,b) = b.flag ? b : (a.s + b.s, a.f|b.f)
// ---------------------------------------------------------------------------
__global__ __launch_bounds__(256) void scan_pass1(
    const float* __restrict__ z, const unsigned char* __restrict__ start,
    float* __restrict__ cs, unsigned char* __restrict__ cf) {
  int idx = blockIdx.x * 256 + threadIdx.x;  // B*H*SNC threads
  int h  = idx & (H_SZ - 1);
  int t2 = idx >> 10;
  int c  = t2 & (SNC - 1);
  int b  = t2 >> 3;
  const float* zp = z + ((size_t)b * T_SZ + (size_t)c * SCH) * H_SZ + h;
  const unsigned char* fp = start + (size_t)b * T_SZ + (size_t)c * SCH;
  float s = 0.0f;
  unsigned char f = 0;
#pragma unroll 8
  for (int i = 0; i < SCH; ++i) {
    float v = zp[(size_t)i * H_SZ];
    if (fp[i]) { s = v; f = 1; } else s += v;
  }
  cs[((size_t)b * H_SZ + h) * SNC + c] = s;
  cf[((size_t)b * H_SZ + h) * SNC + c] = f;
}

__global__ __launch_bounds__(256) void scan_pass2(
    float* __restrict__ cs, const unsigned char* __restrict__ cf) {
  int idx = blockIdx.x * 256 + threadIdx.x;  // B*H threads
  size_t base = (size_t)idx * SNC;
  float e = -6.93147180559945f;  // carry0 = -log(H)
#pragma unroll
  for (int c = 0; c < SNC; ++c) {
    float L = cs[base + c];
    unsigned char F = cf[base + c];
    cs[base + c] = e;              // exclusive prefix carry for chunk c
    e = F ? L : (e + L);
  }
}

__global__ __launch_bounds__(256) void scan_pass3(
    float* __restrict__ z, const unsigned char* __restrict__ start,
    const float* __restrict__ cs) {
  int idx = blockIdx.x * 256 + threadIdx.x;  // B*H*SNC threads
  int h  = idx & (H_SZ - 1);
  int t2 = idx >> 10;
  int c  = t2 & (SNC - 1);
  int b  = t2 >> 3;
  float* zp = z + ((size_t)b * T_SZ + (size_t)c * SCH) * H_SZ + h;
  const unsigned char* fp = start + (size_t)b * T_SZ + (size_t)c * SCH;
  float s = cs[((size_t)b * H_SZ + h) * SNC + c];
#pragma unroll 8
  for (int i = 0; i < SCH; ++i) {
    float v = zp[(size_t)i * H_SZ];
    s = fp[i] ? v : (s + v);
    zp[(size_t)i * H_SZ] = s;
  }
}

// ---------------------------------------------------------------------------
// Row softmax over H=1024, writes bf16 into cat[:, 0:H] (row stride 2H)
// ---------------------------------------------------------------------------
__global__ __launch_bounds__(256) void softmax_kernel(
    const float* __restrict__ st, unsigned short* __restrict__ cat) {
  const int bt  = blockIdx.x;
  const int tid = threadIdx.x;
  const float* row = st + (size_t)bt * H_SZ;
  __shared__ float red[8];
  __shared__ float bmax, bsum;

  float v[4];
  float m = -3.4e38f;
#pragma unroll
  for (int k = 0; k < 4; ++k) {
    v[k] = row[tid + 256 * k];
    m = fmaxf(m, v[k]);
  }
#pragma unroll
  for (int o = 16; o > 0; o >>= 1) m = fmaxf(m, __shfl_xor(m, o, 32));
  if ((tid & 31) == 0) red[tid >> 5] = m;
  __syncthreads();
  if (tid == 0) {
    float mm = red[0];
    for (int i = 1; i < 8; ++i) mm = fmaxf(mm, red[i]);
    bmax = mm;
  }
  __syncthreads();
  m = bmax;

  float s = 0.0f;
#pragma unroll
  for (int k = 0; k < 4; ++k) {
    v[k] = __expf(v[k] - m);
    s += v[k];
  }
#pragma unroll
  for (int o = 16; o > 0; o >>= 1) s += __shfl_xor(s, o, 32);
  if ((tid & 31) == 0) red[tid >> 5] = s;
  __syncthreads();
  if (tid == 0) {
    float ss = 0.0f;
    for (int i = 0; i < 8; ++i) ss += red[i];
    bsum = ss;
  }
  __syncthreads();
  float inv = 1.0f / bsum;

  unsigned short* dst = cat + (size_t)bt * (2 * H_SZ);
#pragma unroll
  for (int k = 0; k < 4; ++k) dst[tid + 256 * k] = f32_to_bf16(v[k] * inv);
}

// ---------------------------------------------------------------------------
extern "C" void kernel_launch(void* const* d_in, const int* in_sizes, int n_in,
                              void* d_out, int out_size, void* d_ws,
                              size_t ws_size, hipStream_t stream) {
  (void)in_sizes; (void)n_in; (void)out_size; (void)ws_size;
  const float*         x     = (const float*)d_in[0];
  const unsigned char* start = (const unsigned char*)d_in[1];
  const float*         Wi    = (const float*)d_in[2];
  const float*         bi    = (const float*)d_in[3];
  const float*         Wp    = (const float*)d_in[4];
  const float*         bp    = (const float*)d_in[5];
  const float*         Wf    = (const float*)d_in[6];
  const float*         bfv   = (const float*)d_in[7];
  const float*         Wo    = (const float*)d_in[8];
  const float*         bo    = (const float*)d_in[9];
  float*               out   = (float*)d_out;

  char* base = (char*)d_ws;
  size_t off = 0;
  auto carve = [&](size_t bytes) -> char* {
    char* p = base + off;
    off = (off + bytes + 255) & ~(size_t)255;
    return p;
  };
  const size_t M = (size_t)B_SZ * T_SZ;  // 32768 rows

  unsigned short* wWi = (unsigned short*)carve((size_t)H_SZ * DIN * 2);
  unsigned short* wWp = (unsigned short*)carve((size_t)NLAYER * H_SZ * H_SZ * 2);
  unsigned short* wWf = (unsigned short*)carve((size_t)NLAYER * H_SZ * 2 * H_SZ * 2);
  unsigned short* wWo = (unsigned short*)carve((size_t)DOUT * H_SZ * 2);
  unsigned short* cat = (unsigned short*)carve(M * 2 * H_SZ * 2);  // [M,2H] bf16
  unsigned short* lay = (unsigned short*)carve(M * H_SZ * 2);      // [M,H]  bf16
  float*          zbf = (float*)carve(M * H_SZ * 4);               // [M,H]  f32
  float*          csb = (float*)carve((size_t)B_SZ * H_SZ * SNC * 4);
  unsigned char*  cfb = (unsigned char*)carve((size_t)B_SZ * H_SZ * SNC);

  dim3 blk(256);

  // weights -> bf16 (packed-pair converts; all sizes are even)
  cvt_bf16_kernel<<<(H_SZ * DIN / 2) / 256, blk, 0, stream>>>(
      Wi, (unsigned int*)wWi, H_SZ * DIN / 2);
  cvt_bf16_kernel<<<(NLAYER * H_SZ * H_SZ / 2) / 256, blk, 0, stream>>>(
      Wp, (unsigned int*)wWp, NLAYER * H_SZ * H_SZ / 2);
  cvt_bf16_kernel<<<(NLAYER * H_SZ * 2 * H_SZ / 2) / 256, blk, 0, stream>>>(
      Wf, (unsigned int*)wWf, NLAYER * H_SZ * 2 * H_SZ / 2);
  cvt_bf16_kernel<<<(DOUT * H_SZ / 2) / 256, blk, 0, stream>>>(
      Wo, (unsigned int*)wWo, DOUT * H_SZ / 2);

  // emb = x @ Wi^T + bi  -> bf16 into cat[:, H:2H] (persistent across layers)
  gemm_wmma_kernel<true, false, false>
      <<<dim3(H_SZ / 128, (int)(M / 128)), blk, 0, stream>>>(
          x, DIN, wWi, bi, cat + H_SZ, 2 * H_SZ, DIN);

  for (int l = 0; l < NLAYER; ++l) {
    const void* Al  = (l == 0) ? (const void*)(cat + H_SZ) : (const void*)lay;
    int         lda = (l == 0) ? 2 * H_SZ : H_SZ;
    // z = layer_in @ Wp[l]^T + bp[l]  (fp32 out for the log-space scan)
    gemm_wmma_kernel<false, false, true>
        <<<dim3(H_SZ / 128, (int)(M / 128)), blk, 0, stream>>>(
            Al, lda, wWp + (size_t)l * H_SZ * H_SZ, bp + l * H_SZ, zbf, H_SZ,
            H_SZ);
    // segmented reset-scan in place on zbf
    scan_pass1<<<(B_SZ * H_SZ * SNC) / 256, blk, 0, stream>>>(zbf, start, csb, cfb);
    scan_pass2<<<(B_SZ * H_SZ) / 256, blk, 0, stream>>>(csb, cfb);
    scan_pass3<<<(B_SZ * H_SZ * SNC) / 256, blk, 0, stream>>>(zbf, start, csb);
    // softmax -> bf16 into cat[:, 0:H]
    softmax_kernel<<<(int)M, blk, 0, stream>>>(zbf, cat);
    // layer_in = leaky_relu(cat @ Wf[l]^T + bf[l]) -> bf16
    gemm_wmma_kernel<false, true, false>
        <<<dim3(H_SZ / 128, (int)(M / 128)), blk, 0, stream>>>(
            cat, 2 * H_SZ, wWf + (size_t)l * H_SZ * 2 * H_SZ, bfv + l * H_SZ,
            lay, H_SZ, 2 * H_SZ);
  }

  // out = layer_in @ Wo^T + bo  (fp32)
  gemm_wmma_kernel<false, false, true>
      <<<dim3(DOUT / 128, (int)(M / 128)), blk, 0, stream>>>(
          lay, H_SZ, wWo, bo, out, DOUT, H_SZ);
}
